// TransformerEncoder_18339510354008
// MI455X (gfx1250) — matrix operand order repaired and checked by reading, not compiled, not run on hardware
//
#include <hip/hip_runtime.h>

typedef unsigned short u16;
typedef __attribute__((ext_vector_type(16))) __bf16 v16bf;
typedef __attribute__((ext_vector_type(8)))  float  v8f;

#define DIM 768
#define HEADS 12
#define HD 64
#define SEQ 1024
#define BATCH 8
#define ROWS (BATCH * SEQ)   // 8192
#define HIDDEN 1536
#define BROW 40              // LDS B-tile row stride in u16 (80B: 16B-aligned, bank-friendly)

__device__ __forceinline__ u16 f2bf(float f) {
    union { float f; unsigned u; } c; c.f = f;
    unsigned u = c.u + 0x7FFFu + ((c.u >> 16) & 1u);
    return (u16)(u >> 16);
}

// A-fragment style: two 16-byte chunks at p0 and p1
__device__ __forceinline__ v16bf load16(const u16* p0, const u16* p1) {
    union { v16bf v; uint4 q[2]; } t;
    t.q[0] = *(const uint4*)p0;
    t.q[1] = *(const uint4*)p1;
    return t.v;
}
// B-fragment style: two 16-byte chunks at p and p+16
__device__ __forceinline__ v16bf load16c(const u16* p) {
    union { v16bf v; uint4 q[2]; } t;
    t.q[0] = *(const uint4*)p;
    t.q[1] = *(const uint4*)(p + 16);
    return t.v;
}

__device__ __forceinline__ v8f wmma_bf16(v16bf a, v16bf b, v8f c) {
    return __builtin_amdgcn_wmma_f32_16x16x32_bf16(
        false, a, false, b, (short)0, c, false, false);
}

// gfx1250 async global->LDS copy (ASYNCcnt-tracked), 16B per lane
__device__ __forceinline__ void async_copy16(const u16* gsrc, unsigned ldsoff) {
    asm volatile("global_load_async_to_lds_b128 %0, %1, off"
                 :: "v"(ldsoff), "v"((unsigned long long)(uintptr_t)gsrc)
                 : "memory");
}
__device__ __forceinline__ void wait_async0() {
    asm volatile("s_wait_asynccnt 0" ::: "memory");
}

// ---------------- weight convert + transpose: W[K x N] f32 -> WT[N x K] bf16
__global__ __launch_bounds__(256)
void wt_kernel(const float* __restrict__ w, u16* __restrict__ wt, int Kd, int Nd) {
    size_t idx = (size_t)blockIdx.x * 256 + threadIdx.x;
    if (idx >= (size_t)Kd * Nd) return;
    int n = (int)(idx % Nd);
    int k = (int)(idx / Nd);
    wt[(size_t)n * Kd + k] = f2bf(w[idx]);
}

// ---------------- layernorm over 768, fp32 in -> bf16 out
__global__ __launch_bounds__(256)
void ln_kernel(const float* __restrict__ x, const float* __restrict__ g,
               const float* __restrict__ be, u16* __restrict__ y) {
    int row = blockIdx.x;
    const float* xr = x + (size_t)row * DIM;
    int t = threadIdx.x;
    float v0 = xr[t], v1 = xr[t + 256], v2 = xr[t + 512];
    float s  = v0 + v1 + v2;
    float ss = v0 * v0 + v1 * v1 + v2 * v2;
    __shared__ float red[18];
    for (int off = 16; off > 0; off >>= 1) {
        s  += __shfl_down(s, off, 32);
        ss += __shfl_down(ss, off, 32);
    }
    int wid = t >> 5;
    if ((t & 31) == 0) { red[wid] = s; red[8 + wid] = ss; }
    __syncthreads();
    if (t == 0) {
        float S = 0.f, SS = 0.f;
        for (int i = 0; i < 8; ++i) { S += red[i]; SS += red[8 + i]; }
        red[16] = S; red[17] = SS;
    }
    __syncthreads();
    float mu   = red[16] * (1.0f / DIM);
    float var  = red[17] * (1.0f / DIM) - mu * mu;
    float rinv = rsqrtf(var + 1e-5f);
    u16* yr = y + (size_t)row * DIM;
    yr[t]       = f2bf((v0 - mu) * rinv * g[t]       + be[t]);
    yr[t + 256] = f2bf((v1 - mu) * rinv * g[t + 256] + be[t + 256]);
    yr[t + 512] = f2bf((v2 - mu) * rinv * g[t + 512] + be[t + 512]);
}

// ---------------- bf16 WMMA GEMM: block = 256 rows x 64 cols, wave = 32x64.
// B tile (64 cols x 32 K) double-buffered in LDS via async global->LDS copies.
enum { EPI_QKV = 0, EPI_RES = 1, EPI_GELU = 2 };

template <int MODE>
__global__ __launch_bounds__(256)
void gemm_kernel(const u16* __restrict__ A, const u16* __restrict__ BT,
                 const float* __restrict__ bias, int M, int N, int K,
                 const float* __restrict__ res, float* __restrict__ outf,
                 u16* __restrict__ outb,
                 u16* __restrict__ Qo, u16* __restrict__ Ko, u16* __restrict__ VTo) {
    __shared__ u16 bs[2][64 * BROW];
    int tid  = threadIdx.x;
    int wv   = tid >> 5, lane = tid & 31;
    int nt64 = N >> 6;
    int mblk = blockIdx.x / nt64;
    int nb   = blockIdx.x % nt64;
    int ln16 = lane & 15, half = lane >> 4;
    int koff = half * 8, kbase = half * 8;

    // per-thread staging slot: thread t stages 16B of row (t>>2), chunk (t&3)
    int sn = tid >> 2, spart = tid & 3;
    const u16* sgsrc   = BT + (size_t)(nb * 64 + sn) * K + spart * 8;
    unsigned   sldsoff = (unsigned)(uintptr_t)(&bs[0][0]) + sn * (BROW * 2) + spart * 16;

    int mrow = mblk * 256 + wv * 32;
    const u16* arow0 = A + (size_t)(mrow + ln16) * K + koff;
    const u16* arow1 = arow0 + (size_t)16 * K;

    v8f acc[8] = {};
    async_copy16(sgsrc, sldsoff);              // prologue: stage k0=0 into buf 0
    unsigned buf = 0;
    for (int k0 = 0; k0 < K; k0 += 32) {
        wait_async0();
        __syncthreads();                        // buf ready for everyone
        if (k0 + 32 < K)
            async_copy16(sgsrc + k0 + 32, sldsoff + (buf ^ 1) * (64 * BROW * 2));
        v16bf a0 = load16(arow0 + k0, arow0 + k0 + 16);
        v16bf a1 = load16(arow1 + k0, arow1 + k0 + 16);
        const u16* bb = &bs[buf][0] + kbase;
#pragma unroll
        for (int nt = 0; nt < 4; ++nt) {
            v16bf b = load16c(bb + (nt * 16 + ln16) * BROW);
            acc[nt]     = wmma_bf16(a0, b, acc[nt]);
            acc[4 + nt] = wmma_bf16(a1, b, acc[4 + nt]);
        }
        buf ^= 1;
    }

#pragma unroll
    for (int h2 = 0; h2 < 2; ++h2) {           // row halves: +0 / +16
#pragma unroll
        for (int nt = 0; nt < 4; ++nt) {
            v8f ac = acc[h2 * 4 + nt];
            int n = nb * 64 + nt * 16 + ln16;
            float bi = bias[n];
#pragma unroll
            for (int r = 0; r < 8; ++r) {
                int gm = mrow + h2 * 16 + r + 8 * half;
                float v = ac[r] + bi;
                if (MODE == EPI_QKV) {
                    int sec = n / DIM, c = n % DIM;
                    int h = c >> 6, d = c & 63;
                    int b = gm >> 10, l = gm & 1023;
                    int bh = b * HEADS + h;
                    if (sec == 0)
                        Qo[((size_t)bh * SEQ + l) * HD + d] = f2bf(v * 0.125f);
                    else if (sec == 1)
                        Ko[((size_t)bh * SEQ + l) * HD + d] = f2bf(v);
                    else
                        VTo[((size_t)bh * HD + d) * SEQ + l] = f2bf(v);
                } else if (MODE == EPI_RES) {
                    outf[(size_t)gm * N + n] = v + res[(size_t)gm * N + n];
                } else { // EPI_GELU (exact)
                    float ge = 0.5f * v * (1.0f + erff(v * 0.70710678118f));
                    outb[(size_t)gm * N + n] = f2bf(ge);
                }
            }
        }
    }
}

// ---------------- flash attention: one wave per (b,h, 16-query tile)
__global__ __launch_bounds__(256)
void attn_kernel(const u16* __restrict__ Q, const u16* __restrict__ Kb,
                 const u16* __restrict__ VT, u16* __restrict__ O) {
    __shared__ u16 smem[8][16 * 32];
    int wv   = threadIdx.x >> 5;
    int lane = threadIdx.x & 31;
    int wave = blockIdx.x * 8 + wv;
    int bh = wave >> 6;        // 0..95
    int qt = wave & 63;        // query tile
    int ln16 = lane & 15, half = lane >> 4;
    int koff = half * 8, kbase = half * 8;

    const u16* qrow = Q + ((size_t)bh * SEQ + qt * 16 + ln16) * HD + koff;
    v16bf aq0 = load16(qrow,       qrow + 16);
    v16bf aq1 = load16(qrow + 32,  qrow + 48);

    v8f o0 = {}, o1 = {}, o2 = {}, o3 = {};
    float runmax[8], runsum[8];
#pragma unroll
    for (int r = 0; r < 8; ++r) { runmax[r] = -1e30f; runsum[r] = 0.f; }
    u16* sp = &smem[wv][0];

    for (int kw = 0; kw < SEQ; kw += 32) {
        // ---- scores: two 16-key tiles, K-dim = 64 (two WMMA each)
        v8f sc0 = {}, sc1 = {};
        {
            const u16* krow = Kb + ((size_t)bh * SEQ + kw + ln16) * HD + kbase;
            sc0 = wmma_bf16(aq0, load16c(krow),      sc0);
            sc0 = wmma_bf16(aq1, load16c(krow + 32), sc0);
            krow += (size_t)16 * HD;
            sc1 = wmma_bf16(aq0, load16c(krow),      sc1);
            sc1 = wmma_bf16(aq1, load16c(krow + 32), sc1);
        }
        // ---- online softmax per row; stash probs (bf16) in LDS
#pragma unroll
        for (int r = 0; r < 8; ++r) {
            float tmax = fmaxf(sc0[r], sc1[r]);
            for (int msk = 8; msk > 0; msk >>= 1)
                tmax = fmaxf(tmax, __shfl_xor(tmax, msk, 32));
            float nm   = fmaxf(runmax[r], tmax);
            float corr = __expf(runmax[r] - nm);
            float p0   = __expf(sc0[r] - nm);
            float p1   = __expf(sc1[r] - nm);
            float ps   = p0 + p1;
            for (int msk = 8; msk > 0; msk >>= 1)
                ps += __shfl_xor(ps, msk, 32);
            runsum[r] = runsum[r] * corr + ps;
            runmax[r] = nm;
            o0[r] *= corr; o1[r] *= corr; o2[r] *= corr; o3[r] *= corr;
            int rr = r + 8 * half;
            sp[rr * 32 + ln16]      = f2bf(p0);
            sp[rr * 32 + 16 + ln16] = f2bf(p1);
        }
        asm volatile("s_wait_dscnt 0" ::: "memory");
        // ---- P (A-frag from LDS) x V (B-frag from VT), 4 d-tiles
        const u16* lp = sp + ln16 * 32 + koff;
        v16bf pa = load16(lp, lp + 16);
        const u16* vrow = VT + ((size_t)bh * HD + ln16) * SEQ + kw + kbase;
        o0 = wmma_bf16(pa, load16c(vrow), o0); vrow += (size_t)16 * SEQ;
        o1 = wmma_bf16(pa, load16c(vrow), o1); vrow += (size_t)16 * SEQ;
        o2 = wmma_bf16(pa, load16c(vrow), o2); vrow += (size_t)16 * SEQ;
        o3 = wmma_bf16(pa, load16c(vrow), o3);
    }

    int b = bh / HEADS, h = bh % HEADS;
#pragma unroll
    for (int r = 0; r < 8; ++r) {
        float inv = 1.0f / runsum[r];
        int gq = qt * 16 + r + 8 * half;
        size_t orow = ((size_t)b * SEQ + gq) * DIM + h * HD;
        O[orow +      ln16] = f2bf(o0[r] * inv);
        O[orow + 16 + ln16] = f2bf(o1[r] * inv);
        O[orow + 32 + ln16] = f2bf(o2[r] * inv);
        O[orow + 48 + ln16] = f2bf(o3[r] * inv);
    }
}

extern "C" void kernel_launch(void* const* d_in, const int* in_sizes, int n_in,
                              void* d_out, int out_size, void* d_ws, size_t ws_size,
                              hipStream_t stream) {
    const float* x      = (const float*)d_in[0];
    const float* ln1_g  = (const float*)d_in[1];
    const float* ln1_b  = (const float*)d_in[2];
    const float* w_qkv  = (const float*)d_in[3];
    const float* b_qkv  = (const float*)d_in[4];
    const float* w_proj = (const float*)d_in[5];
    const float* b_proj = (const float*)d_in[6];
    const float* ln2_g  = (const float*)d_in[7];
    const float* ln2_b  = (const float*)d_in[8];
    const float* w_fc1  = (const float*)d_in[9];
    const float* b_fc1  = (const float*)d_in[10];
    const float* w_fc2  = (const float*)d_in[11];
    const float* b_fc2  = (const float*)d_in[12];
    float* out = (float*)d_out;

    char* ws = (char*)d_ws;
    size_t off = 0;
    auto carve = [&](size_t bytes) { char* p = ws + off; off += (bytes + 255) & ~(size_t)255; return p; };
    u16*   wqkvT  = (u16*)carve((size_t)DIM * 3 * DIM * 2);        // [2304 x 768]
    u16*   wprojT = (u16*)carve((size_t)DIM * DIM * 2);            // [768 x 768]
    u16*   wfc1T  = (u16*)carve((size_t)HIDDEN * DIM * 2);         // [1536 x 768]
    u16*   wfc2T  = (u16*)carve((size_t)DIM * HIDDEN * 2);         // [768 x 1536]
    float* x1     = (float*)carve((size_t)ROWS * DIM * 4);         // post-attn residual
    u16*   lnbuf  = (u16*)carve((size_t)ROWS * DIM * 2);           // ln1 out; reused as O
    u16*   Qb     = (u16*)carve((size_t)ROWS * DIM * 2);           // Q; reused as ln2 out
    u16*   Kbuf   = (u16*)carve((size_t)ROWS * DIM * 2);           // K; Kbuf..VT reused as gelu
    u16*   VTbuf  = (u16*)carve((size_t)ROWS * DIM * 2);           // V transposed
    u16*   Obuf   = lnbuf;
    u16*   h2buf  = Qb;
    u16*   gelu   = Kbuf;     // needs 25.2 MB: spans Kbuf+VTbuf (contiguous)
    (void)ws_size; (void)in_sizes; (void)n_in; (void)out_size;

    // 1) weight transpose+convert
    wt_kernel<<<(DIM * 3 * DIM) / 256, 256, 0, stream>>>(w_qkv,  wqkvT,  DIM,    3 * DIM);
    wt_kernel<<<(DIM * DIM) / 256,     256, 0, stream>>>(w_proj, wprojT, DIM,    DIM);
    wt_kernel<<<(DIM * HIDDEN) / 256,  256, 0, stream>>>(w_fc1,  wfc1T,  DIM,    HIDDEN);
    wt_kernel<<<(HIDDEN * DIM) / 256,  256, 0, stream>>>(w_fc2,  wfc2T,  HIDDEN, DIM);

    // 2) LN1 -> bf16
    ln_kernel<<<ROWS, 256, 0, stream>>>(x, ln1_g, ln1_b, lnbuf);

    // 3) QKV GEMM, scatter Q(scaled)/K/VT
    gemm_kernel<EPI_QKV><<<(ROWS / 256) * (3 * DIM / 64), 256, 0, stream>>>(
        lnbuf, wqkvT, b_qkv, ROWS, 3 * DIM, DIM,
        nullptr, nullptr, nullptr, Qb, Kbuf, VTbuf);

    // 4) attention (flash, one wave per 16-query tile): O overwrites lnbuf
    attn_kernel<<<(BATCH * HEADS * (SEQ / 16)) / 8, 256, 0, stream>>>(Qb, Kbuf, VTbuf, Obuf);

    // 5) proj + residual(x) -> x1 (fp32)
    gemm_kernel<EPI_RES><<<(ROWS / 256) * (DIM / 64), 256, 0, stream>>>(
        Obuf, wprojT, b_proj, ROWS, DIM, DIM,
        x, x1, nullptr, nullptr, nullptr, nullptr);

    // 6) LN2 -> bf16 (into Qb)
    ln_kernel<<<ROWS, 256, 0, stream>>>(x1, ln2_g, ln2_b, h2buf);

    // 7) FC1 + GELU -> bf16 (into Kbuf..VTbuf)
    gemm_kernel<EPI_GELU><<<(ROWS / 256) * (HIDDEN / 64), 256, 0, stream>>>(
        h2buf, wfc1T, b_fc1, ROWS, HIDDEN, DIM,
        nullptr, nullptr, gelu, nullptr, nullptr, nullptr);

    // 8) FC2 + residual(x1) -> out (fp32)
    gemm_kernel<EPI_RES><<<(ROWS / 256) * (DIM / 64), 256, 0, stream>>>(
        gelu, wfc2T, b_fc2, ROWS, DIM, HIDDEN,
        x1, out, nullptr, nullptr, nullptr, nullptr);
}
